// CrossModalAttention_23270132810219
// MI455X (gfx1250) — matrix operand order repaired and checked by reading, not compiled, not run on hardware
//
#include <hip/hip_runtime.h>
#include <cstdint>
#include <cstddef>

// Problem constants (match reference): B=4, T=1024, D=1024, H=16, d=64
#define BB 4
#define TT 1024
#define DD 1024
#define HH 16
#define HD 64
#define BT (BB * TT)   // 4096 rows

typedef __attribute__((ext_vector_type(16))) __bf16 v16bf;
typedef __attribute__((ext_vector_type(8)))  float  v8f;

// ---------- helpers ----------

__device__ __forceinline__ __bf16 f2bf(float f) {
    unsigned int u = __builtin_bit_cast(unsigned int, f);
    unsigned int r = u + 0x7FFFu + ((u >> 16) & 1u);   // round-to-nearest-even
    unsigned short h = (unsigned short)(r >> 16);
    return __builtin_bit_cast(__bf16, h);
}

// 16-bit A-matrix 16x32 element->K map (ISA 7.12.2): elems 0-7 -> K 0-7 (+8*half),
// elems 8-15 -> K 16-23 (+8*half)
__device__ __forceinline__ int kmapA(int j, int hf) {
    return ((j & 8) << 1) + (hf << 3) + (j & 7);
}
// 16-bit B-matrix 32x16 element->K map (same as A): elems 0-7 -> K 0-7 (+8*half),
// elems 8-15 -> K 16-23 (+8*half)
__device__ __forceinline__ int kmapB(int j, int hf) {
    return ((j & 8) << 1) + (hf << 3) + (j & 7);
}

// CDNA5 async global->LDS copy (GLOBAL_LOAD_ASYNC_TO_LDS_B128, tracked by ASYNCcnt).
// LDS operand = low 32 bits of the generic shared pointer (flat->LDS aperture keeps
// the LDS byte offset in addr[31:0]); global address in a 64-bit VGPR pair.
__device__ __forceinline__ void async_g2l_b128(const void* g, void* l) {
    unsigned lo = (unsigned)(uintptr_t)l;
    unsigned long long ga = (unsigned long long)(uintptr_t)g;
    asm volatile("global_load_async_to_lds_b128 %0, %1, off"
                 :: "v"(lo), "v"(ga) : "memory");
}

template <int N>
__device__ __forceinline__ void wait_asynccnt() {
    asm volatile("s_wait_asynccnt %0" :: "i"(N) : "memory");
}

// ---------- elementwise kernels ----------

__global__ void k_f32_to_bf16(const float* __restrict__ in, __bf16* __restrict__ out, int n) {
    int i = blockIdx.x * 256 + threadIdx.x;
    if (i < n) out[i] = f2bf(in[i]);
}

__global__ void k_phase_mean(const float* __restrict__ ph, float* __restrict__ out, int n) {
    int i = blockIdx.x * 256 + threadIdx.x;
    if (i < n) {
        float s = 0.f;
#pragma unroll
        for (int j = 0; j < 16; ++j) s += ph[i * 16 + j];
        out[i] = s * 0.0625f;
    }
}

// ---------- tiled bf16 GEMM: C[M,N] = A[M,K] * B[K,N] ----------
// A,B bf16 row-major. 256 threads (8 waves). 128x64 block tile, 32x32 wave tile
// (2x2 v_wmma_f32_16x16x32_bf16 fragments). Double-buffered LDS fed by async
// global->LDS b128 copies (3 per thread per K-tile); f32 accumulate.

template <bool F32OUT>
__global__ __launch_bounds__(256) void k_gemm128(const __bf16* __restrict__ A,
                                                 const __bf16* __restrict__ B,
                                                 void* __restrict__ Cout,
                                                 int M, int N, int K) {
    __shared__ __align__(16) __bf16 As[2][128 * 32];
    __shared__ __align__(16) __bf16 Bs[2][32 * 64];

    const int m0 = blockIdx.y * 128, n0 = blockIdx.x * 64;
    const int t = threadIdx.x;
    const int wv = t >> 5, lane = t & 31, hf = lane >> 4, idx = lane & 15;
    const int wm = (wv >> 1) * 32, wn = (wv & 1) * 32;   // wave grid 4(M) x 2(N)

    const int arow = t >> 1, acol = (t & 1) * 16;        // A tile: 128 rows x 32 cols
    const int brow = t >> 3, bcol = (t & 7) * 8;         // B tile: 32 rows x 64 cols

    v8f acc[2][2] = {};

    auto stage = [&](int k0, int bi) {
        const __bf16* ag = A + (size_t)(m0 + arow) * K + k0 + acol;
        __bf16* al = &As[bi][arow * 32 + acol];
        async_g2l_b128(ag, al);
        async_g2l_b128(ag + 8, al + 8);
        async_g2l_b128(B + (size_t)(k0 + brow) * N + n0 + bcol,
                       &Bs[bi][brow * 64 + bcol]);
    };

    const int nk = K / 32;
    stage(0, 0);
    for (int ki = 0; ki < nk; ++ki) {
        const int bi = ki & 1;
        if (ki + 1 < nk) {
            stage((ki + 1) * 32, bi ^ 1);   // overlap next tile's copy with compute
            wait_asynccnt<3>();             // this wave's tile-ki copies landed
        } else {
            wait_asynccnt<0>();
        }
        __syncthreads();

        v16bf af[2], bfr[2];
#pragma unroll
        for (int i = 0; i < 2; ++i) {
#pragma unroll
            for (int j = 0; j < 16; ++j) {
                af[i][j]  = As[bi][(wm + i * 16 + idx) * 32 + kmapA(j, hf)];
                bfr[i][j] = Bs[bi][kmapB(j, hf) * 64 + wn + i * 16 + idx];
            }
        }
#pragma unroll
        for (int i = 0; i < 2; ++i)
#pragma unroll
            for (int j = 0; j < 2; ++j)
                acc[i][j] = __builtin_amdgcn_wmma_f32_16x16x32_bf16(
                    false, af[i], false, bfr[j], (short)0, acc[i][j], false, false);
        __syncthreads();   // all waves done reading buffer bi before it is refilled
    }

    // epilogue: C 16x16 f32 layout -> lane idx = N col, vgpr r = M row (m = r + 8*half)
#pragma unroll
    for (int i = 0; i < 2; ++i)
#pragma unroll
        for (int j = 0; j < 2; ++j) {
            const int gmb = m0 + wm + i * 16 + hf * 8;
            const int gn  = n0 + wn + j * 16 + idx;
#pragma unroll
            for (int r = 0; r < 8; ++r) {
                float v = acc[i][j][r];
                if (F32OUT) ((float*)Cout)[(size_t)(gmb + r) * N + gn] = v;
                else        ((__bf16*)Cout)[(size_t)(gmb + r) * N + gn] = f2bf(v);
            }
        }
}

// ---------- phase-gated flash cross-attention ----------
// a = exp(s - m) * gate / sum(exp(s - m) * gate)   (softmax Z cancels after renorm)
// Grid: (T/64, H, B). 128 threads / 4 waves; wave w owns 16 query rows.

__global__ __launch_bounds__(128) void k_attn(const __bf16* __restrict__ Q,
                                              const __bf16* __restrict__ Kg,
                                              const __bf16* __restrict__ Vg,
                                              const float* __restrict__ pq_g,
                                              const float* __restrict__ pk_g,
                                              __bf16* __restrict__ Out) {
    __shared__ __align__(16) __bf16 Qs[64 * 64];
    __shared__ __align__(16) __bf16 Ks[64 * 64];
    __shared__ __align__(16) __bf16 Vs[64 * 64];
    __shared__ __align__(16) __bf16 Ps[4 * 16 * 64];   // per-wave P scratch
    __shared__ float pqs[64];
    __shared__ float pks[64];

    const int b = blockIdx.z, h = blockIdx.y, q0 = blockIdx.x * 64;
    const int t = threadIdx.x;
    const int wv = t >> 5, lane = t & 31, hf = lane >> 4, idx = lane & 15;

    // stage Q tile (64 q-rows x 64 dims) via async copies: 4x b128 per thread
    {
        const int row = t >> 1, ch = (t & 1) * 32;
        const __bf16* src = Q + ((size_t)(b * TT + q0 + row)) * DD + h * HD + ch;
        __bf16* dst = Qs + row * 64 + ch;
#pragma unroll
        for (int q = 0; q < 4; ++q) async_g2l_b128(src + q * 8, dst + q * 8);
    }
    if (t < 64) pqs[t] = pq_g[b * TT + q0 + t];
    wait_asynccnt<0>();
    __syncthreads();

    // Q A-fragments (constant over the Tk loop): rows = wave's 16 q rows, k = dim
    v16bf aq[2];
#pragma unroll
    for (int s = 0; s < 2; ++s)
#pragma unroll
        for (int j = 0; j < 16; ++j)
            aq[s][j] = Qs[(wv * 16 + idx) * 64 + s * 32 + kmapA(j, hf)];

    float pqr[8];
#pragma unroll
    for (int r = 0; r < 8; ++r) pqr[r] = pqs[wv * 16 + r + 8 * hf];

    float rowm[8], rowl[8];
#pragma unroll
    for (int r = 0; r < 8; ++r) { rowm[r] = -3.0e38f; rowl[r] = 0.f; }
    v8f oacc[4] = {};

    for (int kt = 0; kt < TT; kt += 64) {
        // stage K (threads 0-63) and V (threads 64-127): one 128B row each, async
        {
            const int row = t & 63;
            const __bf16* gsrc = ((t < 64) ? Kg : Vg) + ((size_t)(b * TT + kt + row)) * DD + h * HD;
            __bf16* ldst = ((t < 64) ? Ks : Vs) + row * 64;
#pragma unroll
            for (int q = 0; q < 8; ++q) async_g2l_b128(gsrc + q * 8, ldst + q * 8);
        }
        if (t < 64) pks[t] = pk_g[b * TT + kt + t];
        wait_asynccnt<0>();
        __syncthreads();

        // S = Q K^T / sqrt(d): 4 n-fragments of 16 key tokens, 2 k-steps over d=64
        float sv[4][8];
        float tmax[8];
#pragma unroll
        for (int r = 0; r < 8; ++r) tmax[r] = -3.0e38f;

#pragma unroll
        for (int nf = 0; nf < 4; ++nf) {
            v16bf bk[2];
#pragma unroll
            for (int s = 0; s < 2; ++s)
#pragma unroll
                for (int j = 0; j < 16; ++j)
                    bk[s][j] = Ks[(nf * 16 + idx) * 64 + s * 32 + kmapB(j, hf)];
            v8f sc = {};
            sc = __builtin_amdgcn_wmma_f32_16x16x32_bf16(false, aq[0], false, bk[0], (short)0, sc, false, false);
            sc = __builtin_amdgcn_wmma_f32_16x16x32_bf16(false, aq[1], false, bk[1], (short)0, sc, false, false);
#pragma unroll
            for (int r = 0; r < 8; ++r) {
                float s = sc[r] * 0.125f;     // 1/sqrt(64)
                sv[nf][r] = s;
                tmax[r] = fmaxf(tmax[r], s);
            }
        }
        // per-q-row max across the 16 lanes of this half
#pragma unroll
        for (int m_ = 1; m_ < 16; m_ <<= 1)
#pragma unroll
            for (int r = 0; r < 8; ++r)
                tmax[r] = fmaxf(tmax[r], __shfl_xor(tmax[r], m_, 32));

        float corr[8];
#pragma unroll
        for (int r = 0; r < 8; ++r) {
            float mn = fmaxf(rowm[r], tmax[r]);
            corr[r] = __expf(rowm[r] - mn);
            rowm[r] = mn;
            rowl[r] *= corr[r];
        }
#pragma unroll
        for (int f = 0; f < 4; ++f)
#pragma unroll
            for (int r = 0; r < 8; ++r) oacc[f][r] *= corr[r];

        // gated exp weights; write P (C layout -> LDS rows) for A-layout regather
        float lsum[8];
#pragma unroll
        for (int r = 0; r < 8; ++r) lsum[r] = 0.f;
        __bf16* Pw = Ps + wv * 16 * 64;
#pragma unroll
        for (int nf = 0; nf < 4; ++nf) {
            const float pkv = pks[nf * 16 + idx];
#pragma unroll
            for (int r = 0; r < 8; ++r) {
                float g = 0.5f * (1.0f + __cosf(pqr[r] - pkv));
                float wvv = __expf(sv[nf][r] - rowm[r]) * g;
                lsum[r] += wvv;
                Pw[(r + 8 * hf) * 64 + nf * 16 + idx] = f2bf(wvv);
            }
        }
#pragma unroll
        for (int m_ = 1; m_ < 16; m_ <<= 1)
#pragma unroll
            for (int r = 0; r < 8; ++r) lsum[r] += __shfl_xor(lsum[r], m_, 32);
#pragma unroll
        for (int r = 0; r < 8; ++r) rowl[r] += lsum[r];

        // O += P (16x64 tokens) @ V (64 tokens x 64 dims); same-wave LDS RAW is in-order
        v16bf ap[2];
#pragma unroll
        for (int s = 0; s < 2; ++s)
#pragma unroll
            for (int j = 0; j < 16; ++j)
                ap[s][j] = Pw[idx * 64 + s * 32 + kmapA(j, hf)];
#pragma unroll
        for (int f = 0; f < 4; ++f) {
            v16bf bv[2];
#pragma unroll
            for (int s = 0; s < 2; ++s)
#pragma unroll
                for (int j = 0; j < 16; ++j)
                    bv[s][j] = Vs[(s * 32 + kmapB(j, hf)) * 64 + f * 16 + idx];
            oacc[f] = __builtin_amdgcn_wmma_f32_16x16x32_bf16(false, ap[0], false, bv[0], (short)0, oacc[f], false, false);
            oacc[f] = __builtin_amdgcn_wmma_f32_16x16x32_bf16(false, ap[1], false, bv[1], (short)0, oacc[f], false, false);
        }
        __syncthreads();
    }

    // normalize and store attended (bf16) for the output projection GEMM
    float inv[8];
#pragma unroll
    for (int r = 0; r < 8; ++r) inv[r] = 1.0f / (rowl[r] + 1e-8f);
#pragma unroll
    for (int f = 0; f < 4; ++f)
#pragma unroll
        for (int r = 0; r < 8; ++r)
            Out[((size_t)(b * TT + q0 + wv * 16 + 8 * hf + r)) * DD + h * HD + f * 16 + idx] =
                f2bf(oacc[f][r] * inv[r]);
}

// ---------- host orchestration ----------

extern "C" void kernel_launch(void* const* d_in, const int* in_sizes, int n_in,
                              void* d_out, int out_size, void* d_ws, size_t ws_size,
                              hipStream_t stream) {
    (void)in_sizes; (void)n_in; (void)out_size; (void)ws_size;

    const float* x_a  = (const float*)d_in[0];
    const float* x_b  = (const float*)d_in[1];
    const float* ph_a = (const float*)d_in[2];
    const float* ph_b = (const float*)d_in[3];
    const float* Wf[8];
    for (int i = 0; i < 8; ++i) Wf[i] = (const float*)d_in[4 + i];
    // Wf order: 0=W_qa 1=W_kb 2=W_vb 3=W_oa 4=W_qb 5=W_ka 6=W_va 7=W_ob

    char* p = (char*)d_ws;
    auto take = [&](size_t bytes) -> void* {
        void* r = (void*)p;
        p += (bytes + 255) & ~(size_t)255;
        return r;
    };

    __bf16* Xa = (__bf16*)take((size_t)BT * DD * 2);
    __bf16* Xb = (__bf16*)take((size_t)BT * DD * 2);
    __bf16* W16[8];
    for (int i = 0; i < 8; ++i) W16[i] = (__bf16*)take((size_t)DD * DD * 2);
    __bf16* Qa = (__bf16*)take((size_t)BT * DD * 2);
    __bf16* Ka = (__bf16*)take((size_t)BT * DD * 2);
    __bf16* Va = (__bf16*)take((size_t)BT * DD * 2);
    __bf16* Qb = (__bf16*)take((size_t)BT * DD * 2);
    __bf16* Kb = (__bf16*)take((size_t)BT * DD * 2);
    __bf16* Vb = (__bf16*)take((size_t)BT * DD * 2);
    __bf16* Atta = (__bf16*)take((size_t)BT * DD * 2);
    __bf16* Attb = (__bf16*)take((size_t)BT * DD * 2);
    float* pqa = (float*)take((size_t)BT * 4);
    float* pqb = (float*)take((size_t)BT * 4);

    // 1) downconvert inputs
    k_f32_to_bf16<<<(BT * DD) / 256, 256, 0, stream>>>(x_a, Xa, BT * DD);
    k_f32_to_bf16<<<(BT * DD) / 256, 256, 0, stream>>>(x_b, Xb, BT * DD);
    for (int i = 0; i < 8; ++i)
        k_f32_to_bf16<<<(DD * DD) / 256, 256, 0, stream>>>(Wf[i], W16[i], DD * DD);
    k_phase_mean<<<BT / 256, 256, 0, stream>>>(ph_a, pqa, BT);
    k_phase_mean<<<BT / 256, 256, 0, stream>>>(ph_b, pqb, BT);

    // 2) QKV projections (bf16 out)
    dim3 ggrid(DD / 64, BT / 128);
    dim3 gb(256);
    k_gemm128<false><<<ggrid, gb, 0, stream>>>(Xa, W16[0], Qa, BT, DD, DD);  // q_a
    k_gemm128<false><<<ggrid, gb, 0, stream>>>(Xb, W16[1], Kb, BT, DD, DD);  // k_b
    k_gemm128<false><<<ggrid, gb, 0, stream>>>(Xb, W16[2], Vb, BT, DD, DD);  // v_b
    k_gemm128<false><<<ggrid, gb, 0, stream>>>(Xb, W16[4], Qb, BT, DD, DD);  // q_b
    k_gemm128<false><<<ggrid, gb, 0, stream>>>(Xa, W16[5], Ka, BT, DD, DD);  // k_a
    k_gemm128<false><<<ggrid, gb, 0, stream>>>(Xa, W16[6], Va, BT, DD, DD);  // v_a

    // 3) phase-gated flash cross-attention, both directions
    dim3 agrid(TT / 64, HH, BB);
    dim3 ab(128);
    k_attn<<<agrid, ab, 0, stream>>>(Qa, Kb, Vb, pqa, pqb, Atta);
    k_attn<<<agrid, ab, 0, stream>>>(Qb, Ka, Va, pqb, pqa, Attb);

    // 4) output projections straight into d_out (f32), attended_a then attended_b
    float* out = (float*)d_out;
    k_gemm128<true><<<ggrid, gb, 0, stream>>>(Atta, W16[3], out, BT, DD, DD);
    k_gemm128<true><<<ggrid, gb, 0, stream>>>(Attb, W16[7], out + (size_t)BT * DD, BT, DD, DD);
}
